// SelfAttentionLayer_71038759075840
// MI455X (gfx1250) — matrix operand order repaired and checked by reading, not compiled, not run on hardware
//
#include <hip/hip_runtime.h>
#include <hip/hip_bf16.h>
#include <math.h>

typedef __attribute__((ext_vector_type(16))) __bf16 v16bf;
typedef __attribute__((ext_vector_type(8)))  __bf16 v8bf;
typedef __attribute__((ext_vector_type(8)))  float  v8f;

// Exact pointee type expected by __builtin_amdgcn_global_load_async_to_lds_b128
// (from the compiler diagnostic): int __attribute__((vector_size(16))).
typedef int v4i_t __attribute__((vector_size(4 * sizeof(int))));
typedef __attribute__((address_space(1))) v4i_t* gv4i_p;   // global (prints as __device__)
typedef __attribute__((address_space(3))) v4i_t* lv4i_p;   // LDS    (prints as __shared__)

static constexpr int B = 4, N = 4096, C = 256;
static constexpr float LN_EPS = 1e-6f;
static constexpr float STAB   = 1e-9f;
static constexpr float NEG_BIG = -3.0e38f;

#if __has_builtin(__builtin_amdgcn_global_load_async_to_lds_b128) && \
    __has_builtin(__builtin_amdgcn_s_wait_asynccnt)
#define USE_ASYNC_LDS 1
#else
#define USE_ASYNC_LDS 0
#endif

// ---------------- WMMA helpers (CDNA5 16x16x32 bf16, f32 accum) ----------------

__device__ __forceinline__ v8f wmma_bf(v16bf a, v16bf b, v8f c) {
  // (neg_a, A, neg_b, B, c_mod, C, reuse_a, reuse_b)
  return __builtin_amdgcn_wmma_f32_16x16x32_bf16(false, a, false, b, (short)0, c, false, false);
}

// A fragment: 16x32 (MxK) bf16, row-major source, ld = row stride (elements).
// Lanes 0-15 row M=lane hold K=0..7 / K=16..23; lanes 16-31 hold K=8..15 / 24..31.
__device__ __forceinline__ v16bf load_afrag(const __bf16* base, int ld, int mrow, int kc) {
  const int lo = threadIdx.x & 15, hi = (threadIdx.x >> 4) & 1;
  const __bf16* p = base + (size_t)(mrow + lo) * ld + kc + hi * 8;
  v8bf a0 = *reinterpret_cast<const v8bf*>(p);
  v8bf a1 = *reinterpret_cast<const v8bf*>(p + 16);
  v16bf r;
#pragma unroll
  for (int i = 0; i < 8; ++i) { r[i] = a0[i]; r[i + 8] = a1[i]; }
  return r;
}

// B fragment: 32x16 (KxN). Operand column n = contiguous row n of the source
// (source stored [N, K] row-major: k-matrix rows / transposed weights).
__device__ __forceinline__ v16bf load_bfrag(const __bf16* base, int ld, int nrow, int kc) {
  const int lo = threadIdx.x & 15, hi = (threadIdx.x >> 4) & 1;
  const __bf16* p = base + (size_t)(nrow + lo) * ld + kc + hi * 8;
  v8bf b0 = *reinterpret_cast<const v8bf*>(p);
  v8bf b1 = *reinterpret_cast<const v8bf*>(p + 16);
  v16bf r;
#pragma unroll
  for (int i = 0; i < 8; ++i) { r[i] = b0[i]; r[i + 8] = b1[i]; }
  return r;
}

// ---------------- LDS tile staging (8KB: 16 rows x 256 bf16, contiguous) ----------------
// 256 threads, 32B each. Async-to-LDS when available (ASYNCcnt), else ld+ds_store.

__device__ __forceinline__ void stage_tile(const __bf16* __restrict__ g, __bf16* l) {
  const int t = threadIdx.x;
#if USE_ASYNC_LDS
  gv4i_p gp = (gv4i_p)g;                 // addrspacecast + drop const + retype to int4
  lv4i_p lp = (lv4i_p)l;
  __builtin_amdgcn_global_load_async_to_lds_b128(gp + t * 2,     lp + t * 2,     0, 0);
  __builtin_amdgcn_global_load_async_to_lds_b128(gp + t * 2 + 1, lp + t * 2 + 1, 0, 0);
#else
  const uint4* gs = (const uint4*)g;
  uint4* ls = (uint4*)l;
  ls[t * 2]     = gs[t * 2];
  ls[t * 2 + 1] = gs[t * 2 + 1];
#endif
}

__device__ __forceinline__ void wait_stage() {
#if USE_ASYNC_LDS
  __builtin_amdgcn_s_wait_asynccnt(0);
#endif
  __syncthreads();
}

// ---------------- prep kernels ----------------

__global__ void k_cvt_bf16(const float* __restrict__ src, __bf16* __restrict__ dst, int n) {
  int i = blockIdx.x * blockDim.x + threadIdx.x;
  if (i < n) dst[i] = (__bf16)src[i];
}

__global__ void k_wT(const float* __restrict__ wq, const float* __restrict__ wk,
                     const float* __restrict__ wv, const float* __restrict__ wo,
                     __bf16* __restrict__ tq, __bf16* __restrict__ tk,
                     __bf16* __restrict__ tv, __bf16* __restrict__ to_) {
  int i = blockIdx.x * blockDim.x + threadIdx.x;
  if (i >= C * C) return;
  int j = i / C, c = i % C;          // Wt[j, c] = W[c, j]
  tq[i]  = (__bf16)wq[c * C + j];
  tk[i]  = (__bf16)wk[c * C + j];
  tv[i]  = (__bf16)wv[c * C + j];
  to_[i] = (__bf16)wo[c * C + j];
}

__global__ void k_zero(float* __restrict__ s, float* __restrict__ cs,
                       float* __restrict__ cq, float* __restrict__ mc) {
  int i = blockIdx.x * blockDim.x + threadIdx.x;
  if (i < B * N) s[i] = 0.f;
  if (i < B * C) { cs[i] = 0.f; cq[i] = 0.f; }
  if (i < B)     mc[i] = 0.f;
}

__global__ void k_maskcnt(const unsigned char* __restrict__ mask, float* __restrict__ mc) {
  int b = blockIdx.x;
  int cnt = 0;
  for (int n = threadIdx.x; n < N; n += blockDim.x) cnt += (mask[b * N + n] == 0);
  atomicAdd(&mc[b], (float)cnt);
}

// ---------------- fused QKV projection: q,k -> bf16, v -> f32 ----------------

__global__ void k_qkv(const __bf16* __restrict__ xb,
                      const __bf16* __restrict__ tq, const __bf16* __restrict__ tk,
                      const __bf16* __restrict__ tv,
                      const float* __restrict__ bq, const float* __restrict__ bk,
                      const float* __restrict__ bv,
                      __bf16* __restrict__ qb, __bf16* __restrict__ kb,
                      float* __restrict__ vmat) {
  const int wave = (blockIdx.x * blockDim.x + threadIdx.x) >> 5;  // (B*N/16)*(C/16) waves
  const int mt = wave / (C / 16);
  const int jt = wave % (C / 16);
  const int lo = threadIdx.x & 15, hi = (threadIdx.x >> 4) & 1;
  v8f cq = {}, ck = {}, cv = {};
#pragma unroll
  for (int kc = 0; kc < C; kc += 32) {
    v16bf a = load_afrag(xb, C, mt * 16, kc);
    cq = wmma_bf(a, load_bfrag(tq, C, jt * 16, kc), cq);
    ck = wmma_bf(a, load_bfrag(tk, C, jt * 16, kc), ck);
    cv = wmma_bf(a, load_bfrag(tv, C, jt * 16, kc), cv);
  }
  const int col = jt * 16 + lo;
  const float bqv = bq[col], bkv = bk[col], bvv = bv[col];
  const int row0 = mt * 16 + hi * 8;
#pragma unroll
  for (int r = 0; r < 8; ++r) {
    size_t idx = (size_t)(row0 + r) * C + col;
    qb[idx]   = (__bf16)(cq[r] + bqv);
    kb[idx]   = (__bf16)(ck[r] + bkv);
    vmat[idx] = cv[r] + bvv;
  }
}

// ---------------- pass 1: per-row online softmax stats (flash-style) ----------------
// 8 waves/block own 8 consecutive m-tiles (same batch); the streamed k-tile is
// staged once per block through double-buffered LDS (async copies on ASYNCcnt).

__global__ void k_rowstats(const __bf16* __restrict__ qb, const __bf16* __restrict__ kb,
                           const unsigned char* __restrict__ mask,
                           float* __restrict__ rowmax, float* __restrict__ rowsum) {
  __shared__ __bf16 kt[2][16 * C];                 // 2 x 8KB
  const int wv = threadIdx.x >> 5;                 // wave in block: 0..7
  const int mt0 = blockIdx.x * 8;                  // 8 m-tiles per block, batch-aligned
  const int b = mt0 / (N / 16);
  const int grow = (mt0 + wv) * 16;                // this wave's global q-row base
  const int lo = threadIdx.x & 15, hi = (threadIdx.x >> 4) & 1;

  stage_tile(kb + (size_t)(b * N) * C, &kt[0][0]); // prefetch nt = 0

  v16bf afr[8];                                    // whole q row-block in registers
#pragma unroll
  for (int k = 0; k < 8; ++k) afr[k] = load_afrag(qb, C, grow, k * 32);

  float rmax[8], rsum[8];
#pragma unroll
  for (int r = 0; r < 8; ++r) { rmax[r] = NEG_BIG; rsum[r] = 0.f; }

  for (int nt = 0; nt < N / 16; ++nt) {
    const int cur = nt & 1;
    wait_stage();                                  // s_wait_asynccnt 0 + barrier
    if (nt + 1 < N / 16)
      stage_tile(kb + (size_t)(b * N + (nt + 1) * 16) * C, &kt[cur ^ 1][0]);

    v8f acc = {};
#pragma unroll
    for (int k = 0; k < 8; ++k)
      acc = wmma_bf(afr[k], load_bfrag(&kt[cur][0], C, 0, k * 32), acc);

    const bool cm = mask[b * N + nt * 16 + lo] != 0;   // this lane's attention column
#pragma unroll
    for (int r = 0; r < 8; ++r) {
      float a = cm ? acc[r] : NEG_BIG;
      float tm = a;                                // butterfly max within 16-lane half
      tm = fmaxf(tm, __shfl_xor(tm, 1, 32));
      tm = fmaxf(tm, __shfl_xor(tm, 2, 32));
      tm = fmaxf(tm, __shfl_xor(tm, 4, 32));
      tm = fmaxf(tm, __shfl_xor(tm, 8, 32));
      float newm = fmaxf(rmax[r], tm);
      float p = cm ? __expf(a - newm) : 0.f;
      p += __shfl_xor(p, 1, 32);
      p += __shfl_xor(p, 2, 32);
      p += __shfl_xor(p, 4, 32);
      p += __shfl_xor(p, 8, 32);
      rsum[r] = rsum[r] * __expf(rmax[r] - newm) + p;
      rmax[r] = newm;
    }
  }
#pragma unroll
  for (int r = 0; r < 8; ++r) {
    if (lo == r) {
      rowmax[grow + hi * 8 + r] = rmax[r];
      rowsum[grow + hi * 8 + r] = rsum[r];
    }
  }
}

// ---------------- pass 2: column sums of softmax ----------------
// 8 waves/block own 8 consecutive n-tiles (k-fragments register-resident);
// the streamed q-tile goes through double-buffered LDS.

__global__ void k_colsum(const __bf16* __restrict__ qb, const __bf16* __restrict__ kb,
                         const unsigned char* __restrict__ mask,
                         const float* __restrict__ rowmax, const float* __restrict__ rowsum,
                         float* __restrict__ Scol) {
  __shared__ __bf16 qt[2][16 * C];                 // 2 x 8KB
  const int wv = threadIdx.x >> 5;
  const int nt0 = blockIdx.x * 8;
  const int b = nt0 / (N / 16);
  const int ncol = (nt0 + wv) * 16;                // this wave's global key-row base
  const int lo = threadIdx.x & 15, hi = (threadIdx.x >> 4) & 1;

  stage_tile(qb + (size_t)(b * N) * C, &qt[0][0]); // prefetch mt = 0

  v16bf kfr[8];                                    // k column-block in registers
#pragma unroll
  for (int k = 0; k < 8; ++k) kfr[k] = load_bfrag(kb, C, ncol, k * 32);
  const bool cm = mask[ncol + lo] != 0;

  float acc = 0.f;
  for (int mt = 0; mt < N / 16; ++mt) {
    const int cur = mt & 1;
    wait_stage();
    if (mt + 1 < N / 16)
      stage_tile(qb + (size_t)(b * N + (mt + 1) * 16) * C, &qt[cur ^ 1][0]);

    v8f t = {};
#pragma unroll
    for (int k = 0; k < 8; ++k)
      t = wmma_bf(load_afrag(&qt[cur][0], C, 0, k * 32), kfr[k], t);

    const int mrow = b * N + mt * 16;
#pragma unroll
    for (int r = 0; r < 8; ++r) {
      const int m = mrow + hi * 8 + r;
      const bool rm = mask[m] != 0;
      float val = __expf(t[r] - rowmax[m]) / rowsum[m];
      acc += (rm && cm) ? val : 0.f;
    }
  }
  acc += __shfl_xor(acc, 16, 32);                  // combine halves (rows 0-7 / 8-15)
  if (hi == 0) Scol[ncol + lo] = acc;
}

// ---------------- gate: out_pre = v * S/(STAB+S), packed to bf16 for the Wo GEMM --------

__global__ void k_gate(const float* __restrict__ vmat, const float* __restrict__ Scol,
                       const float* __restrict__ mc, __bf16* __restrict__ gb) {
  int i = blockIdx.x * blockDim.x + threadIdx.x;
  if (i >= B * N * C) return;
  int bn = i / C;
  int b  = bn / N;
  float s = Scol[bn] + mc[b] * (1.0f / N);    // masked rows add uniform 1/N everywhere
  float g = s / (STAB + s);
  gb[i] = (__bf16)(vmat[i] * g);
}

// ---------------- output projection + LayerNorm stats over the N axis ----------------

__global__ void k_out(const __bf16* __restrict__ gb, const __bf16* __restrict__ to_,
                      const float* __restrict__ bo, float* __restrict__ y,
                      float* __restrict__ csum, float* __restrict__ csq) {
  const int wave = (blockIdx.x * blockDim.x + threadIdx.x) >> 5;
  const int mt = wave / (C / 16);
  const int jt = wave % (C / 16);
  const int lo = threadIdx.x & 15, hi = (threadIdx.x >> 4) & 1;
  v8f cy = {};
#pragma unroll
  for (int kc = 0; kc < C; kc += 32)
    cy = wmma_bf(load_afrag(gb, C, mt * 16, kc), load_bfrag(to_, C, jt * 16, kc), cy);
  const int col  = jt * 16 + lo;
  const float bias = bo[col];
  const int row0 = mt * 16 + hi * 8;
  const int b    = (mt * 16) / N;
  float s1 = 0.f, s2 = 0.f;
#pragma unroll
  for (int r = 0; r < 8; ++r) {
    float v = cy[r] + bias;
    y[(size_t)(row0 + r) * C + col] = v;
    s1 += v; s2 += v * v;
  }
  s1 += __shfl_xor(s1, 16, 32);
  s2 += __shfl_xor(s2, 16, 32);
  if (hi == 0) {
    atomicAdd(&csum[b * C + col], s1);
    atomicAdd(&csq [b * C + col], s2);
  }
}

// ---------------- finalize: LN (over points axis) + ReLU + residual ----------------

__global__ void k_final(const float* __restrict__ y, const float* __restrict__ csum,
                        const float* __restrict__ csq, const float* __restrict__ ln_s,
                        const float* __restrict__ ln_b, const float* __restrict__ x,
                        float* __restrict__ out) {
  int i = blockIdx.x * blockDim.x + threadIdx.x;
  if (i >= B * N * C) return;
  int c = i % C;
  int b = i / (N * C);
  float mean = csum[b * C + c] * (1.0f / N);
  float var  = csq [b * C + c] * (1.0f / N) - mean * mean;
  float o = (y[i] - mean) * rsqrtf(var + LN_EPS) * ln_s[c] + ln_b[c];
  out[i] = fmaxf(o, 0.f) + x[i];
}

// ---------------- host launch ----------------

extern "C" void kernel_launch(void* const* d_in, const int* in_sizes, int n_in,
                              void* d_out, int out_size, void* d_ws, size_t ws_size,
                              hipStream_t stream) {
  (void)in_sizes; (void)n_in; (void)out_size; (void)ws_size;
  const float*         x    = (const float*)d_in[0];
  const unsigned char* mask = (const unsigned char*)d_in[1];
  const float* Wq = (const float*)d_in[2];
  const float* bq = (const float*)d_in[3];
  const float* Wk = (const float*)d_in[4];
  const float* bk = (const float*)d_in[5];
  const float* Wv = (const float*)d_in[6];
  const float* bv = (const float*)d_in[7];
  const float* Wo = (const float*)d_in[8];
  const float* bo = (const float*)d_in[9];
  const float* ln_s = (const float*)d_in[10];
  const float* ln_b = (const float*)d_in[11];
  float* out = (float*)d_out;

  char* w = (char*)d_ws;
  auto alloc = [&](size_t bytes) {
    char* p = w; w += (bytes + 255) & ~size_t(255); return (void*)p;
  };
  __bf16* xb   = (__bf16*)alloc((size_t)B * N * C * 2);
  __bf16* qb   = (__bf16*)alloc((size_t)B * N * C * 2);
  __bf16* kb   = (__bf16*)alloc((size_t)B * N * C * 2);
  __bf16* gb   = (__bf16*)alloc((size_t)B * N * C * 2);
  __bf16* tq   = (__bf16*)alloc((size_t)C * C * 2);
  __bf16* tk   = (__bf16*)alloc((size_t)C * C * 2);
  __bf16* tv   = (__bf16*)alloc((size_t)C * C * 2);
  __bf16* to_  = (__bf16*)alloc((size_t)C * C * 2);
  float*  vmat = (float*)alloc((size_t)B * N * C * 4);
  float*  y    = (float*)alloc((size_t)B * N * C * 4);
  float*  rmax = (float*)alloc((size_t)B * N * 4);
  float*  rsum = (float*)alloc((size_t)B * N * 4);
  float*  Scol = (float*)alloc((size_t)B * N * 4);
  float*  csum = (float*)alloc((size_t)B * C * 4);
  float*  csq  = (float*)alloc((size_t)B * C * 4);
  float*  mc   = (float*)alloc((size_t)B * 4);

  const int T = 256;                              // 8 waves / block
  const int nEl   = B * N * C;                    // 4.19M
  const int gEl   = (nEl + T - 1) / T;
  const int gWT   = (C * C + T - 1) / T;
  const int gZero = (B * N + T - 1) / T;
  const int gTile = ((B * N / 16) * (C / 16)) / (T / 32);   // 2048 blocks (1 wave = 1 tile)
  const int gRow  = (B * N / 16) / (T / 32);                // 128 blocks (8 tiles / block)

  k_zero    <<<gZero, T, 0, stream>>>(Scol, csum, csq, mc);
  k_cvt_bf16<<<gEl,   T, 0, stream>>>(x, xb, nEl);
  k_wT      <<<gWT,   T, 0, stream>>>(Wq, Wk, Wv, Wo, tq, tk, tv, to_);
  k_maskcnt <<<B,     T, 0, stream>>>(mask, mc);
  k_qkv     <<<gTile, T, 0, stream>>>(xb, tq, tk, tv, bq, bk, bv, qb, kb, vmat);
  k_rowstats<<<gRow,  T, 0, stream>>>(qb, kb, mask, rmax, rsum);
  k_colsum  <<<gRow,  T, 0, stream>>>(qb, kb, mask, rmax, rsum, Scol);
  k_gate    <<<gEl,   T, 0, stream>>>(vmat, Scol, mc, gb);
  k_out     <<<gTile, T, 0, stream>>>(gb, to_, bo, y, csum, csq);
  k_final   <<<gEl,   T, 0, stream>>>(y, csum, csq, ln_s, ln_b, x, out);
}
